// CrossDomainGATLayer_58402965291284
// MI455X (gfx1250) — matrix operand order repaired and checked by reading, not compiled, not run on hardware
//
#include <hip/hip_runtime.h>
#include <hip/hip_bf16.h>
#include <math.h>

// ---------------------------------------------------------------------------
// CDNA5 (gfx1250) WMMA types
// ---------------------------------------------------------------------------
typedef __attribute__((ext_vector_type(16))) _Float16 v16h;
typedef __attribute__((ext_vector_type(8)))  float    v8f;

#define DIMC  256
#define HEADS 8
#define HDIM  32
#define TB    256

// ---------------------------------------------------------------------------
// Pack x[N,256] (f32, row-major) into f16 A-operand panels.
// Panel layout per (m_tile, k_step): 32 lanes x 16 halves, matching the ISA
// 16-bit A-matrix 16x32 layout:
//   lanes 0-15  : row M=lane,    halves 0-7 -> K=0..7,  halves 8-15 -> K=16..23
//   lanes 16-31 : row M=lane-16, halves 0-7 -> K=8..15, halves 8-15 -> K=24..31
// Rows >= Nrows are zero-padded.
// ---------------------------------------------------------------------------
__global__ void pack_a_kernel(const float* __restrict__ x, _Float16* __restrict__ out,
                              int Nrows, int m_tiles) {
    int gid = blockIdx.x * blockDim.x + threadIdx.x;
    int total = m_tiles << 12;                 // m_tiles * 8 ks * 32 lanes * 16 halves
    if (gid >= total) return;
    int i    = gid & 15;
    int lane = (gid >> 4) & 31;
    int ks   = (gid >> 9) & 7;
    int mt   = gid >> 12;
    int row  = mt * 16 + (lane & 15);
    int kk   = (i < 8 ? i : i + 8) + ((lane >> 4) << 3);
    int col  = ks * 32 + kk;
    float v  = (row < Nrows) ? x[(size_t)row * DIMC + col] : 0.0f;
    out[gid] = (_Float16)v;
}

// ---------------------------------------------------------------------------
// Pack W[256,256] (f32, row-major, W[k][n]) into f16 B-operand panels.
// Panel layout per (n_tile, k_step): 32 lanes x 16 halves, per ISA B layout:
//   lane l -> column N = l&15 ; half i -> K = (i<8 ? i : i+8) + 8*(l>>4)
// ---------------------------------------------------------------------------
__global__ void pack_b_kernel(const float* __restrict__ W, _Float16* __restrict__ out) {
    int gid = blockIdx.x * blockDim.x + threadIdx.x;   // 16 nt * 8 ks * 32 * 16 = 65536
    if (gid >= 65536) return;
    int i    = gid & 15;
    int lane = (gid >> 4) & 31;
    int ks   = (gid >> 9) & 7;
    int nt   = gid >> 12;
    int n    = nt * 16 + (lane & 15);
    int k    = ks * 32 + (i < 8 ? i : i + 8) + ((lane >> 4) << 3);
    out[gid] = (_Float16)W[(size_t)k * DIMC + n];
}

// ---------------------------------------------------------------------------
// WMMA GEMM v3: out[N,256] = A[N,256] * W[256,256] + bias (+ resid)
//
// Block = 256 threads = 8 waves. Block stages the ENTIRE packed B (128 KB)
// plus the bias vector (1 KB) into LDS once. Wave w owns m-tile
// blockIdx.x*8+w: A stripe hoisted to 64 VGPRs, then a fully unrolled sweep
// of 16 n-tiles; per tile all 8 B operands are ds_load'ed into a fresh b[8]
// before the 8-deep WMMA chain so LDS latency is covered by one wait and the
// chain runs back-to-back. Stores take a wave-uniform full-tile fast path
// (row guards only on the one ragged tail tile).
// ---------------------------------------------------------------------------
template<bool RESID>
__global__ void gemm_wmma_kernel(const _Float16* __restrict__ Apack,
                                 const _Float16* __restrict__ Bpack,
                                 const float* __restrict__ bias,
                                 const float* __restrict__ resid,
                                 float* __restrict__ out,
                                 int Nrows, int m_tiles) {
    __shared__ _Float16 Bsh[65536];            // 128 KB of the 320 KB WGP LDS
    __shared__ float    BiasSh[DIMC];          // 1 KB

    // ---- cooperative global -> LDS fill (all threads) ----
    {
        const uint4* gsrc = (const uint4*)Bpack;
        uint4*       ldst = (uint4*)Bsh;
#pragma unroll
        for (int t = 0; t < 32; ++t)           // 8192 uint4 / 256 threads
            ldst[t * TB + threadIdx.x] = gsrc[t * TB + threadIdx.x];
        BiasSh[threadIdx.x] = bias[threadIdx.x];
    }
    __syncthreads();

    int wave = threadIdx.x >> 5;
    int lane = threadIdx.x & 31;
    int mt   = blockIdx.x * (blockDim.x >> 5) + wave;
    if (mt >= m_tiles) return;                 // wave-uniform, after the barrier

    // ---- hoist the wave's A stripe into registers (64 VGPRs) ----
    const v16h* Ap = (const v16h*)Apack + (size_t)mt * 256;   // 8 ks * 32 lanes
    v16h a[8];
#pragma unroll
    for (int ks = 0; ks < 8; ++ks)
        a[ks] = Ap[ks * 32 + lane];

    const v16h* Bl    = (const v16h*)Bsh;
    int  ncol0        = lane & 15;
    int  rbase        = mt * 16 + ((lane >> 4) << 3);
    bool fullTile     = (mt * 16 + 16 <= Nrows);   // wave-uniform

    // ---- fully unrolled sweep over the 16 n-tiles, B from LDS ----
#pragma unroll
    for (int nt = 0; nt < 16; ++nt) {
        v16h b[8];
#pragma unroll
        for (int ks = 0; ks < 8; ++ks)
            b[ks] = Bl[(nt * 8 + ks) * 32 + lane];

        int   ncol = nt * 16 + ncol0;
        float bval = BiasSh[ncol];
        v8f c = {bval, bval, bval, bval, bval, bval, bval, bval};
#pragma unroll
        for (int ks = 0; ks < 8; ++ks)
            c = __builtin_amdgcn_wmma_f32_16x16x32_f16(
                    /*neg_a=*/false, a[ks], /*neg_b=*/false, b[ks],
                    /*c_mod=*/(short)0, c, /*reuse_a=*/false, /*reuse_b=*/false);

        // D layout: lanes 0-15 VGPR r -> M=r ; lanes 16-31 VGPR r -> M=8+r
        if (fullTile) {
#pragma unroll
            for (int r = 0; r < 8; ++r) {
                size_t idx = (size_t)(rbase + r) * DIMC + ncol;
                float  v   = c[r];
                if (RESID) v += resid[idx];
                out[idx] = v;
            }
        } else {
#pragma unroll
            for (int r = 0; r < 8; ++r) {
                int row = rbase + r;
                if (row < Nrows) {
                    size_t idx = (size_t)row * DIMC + ncol;
                    float  v   = c[r];
                    if (RESID) v += resid[idx];
                    out[idx] = v;
                }
            }
        }
    }
}

// ---------------------------------------------------------------------------
// Ordered-uint mapping so unsigned atomicMax implements float max (incl. -inf)
// ---------------------------------------------------------------------------
__device__ __forceinline__ unsigned orderF(float f) {
    unsigned u = __float_as_uint(f);
    return (u & 0x80000000u) ? ~u : (u | 0x80000000u);
}
__device__ __forceinline__ float unorderF(unsigned u) {
    unsigned b = (u & 0x80000000u) ? (u ^ 0x80000000u) : ~u;
    return __uint_as_float(b);
}

// init: mU = order(-inf) = 0x007FFFFF, sums = 0, msg = 0
__global__ void init_branch_kernel(unsigned* __restrict__ mU, float* __restrict__ ssum,
                                   float* __restrict__ msg, int NH, int NC) {
    int gid = blockIdx.x * blockDim.x + threadIdx.x;
    if (gid < NC) msg[gid] = 0.0f;
    if (gid < NH) { mU[gid] = 0x007FFFFFu; ssum[gid] = 0.0f; }
}

// scores[e,h] = dot(Q[dst,h,:], K[src,h,:]) / sqrt(32); fused segment-max
__global__ void edge_scores_kernel(const float* __restrict__ Q, const float* __restrict__ K,
                                   const int* __restrict__ src, const int* __restrict__ dst,
                                   float* __restrict__ score, unsigned* __restrict__ mU, int EH) {
    int gid = blockIdx.x * blockDim.x + threadIdx.x;
    if (gid >= EH) return;
    int e = gid >> 3, h = gid & 7;
    int s = src[e], d = dst[e];
    const float4* q = (const float4*)(Q + (size_t)d * DIMC + h * HDIM);
    const float4* k = (const float4*)(K + (size_t)s * DIMC + h * HDIM);
    float acc = 0.0f;
#pragma unroll
    for (int i = 0; i < 8; ++i) {
        float4 a = q[i], b = k[i];
        acc += a.x * b.x + a.y * b.y + a.z * b.z + a.w * b.w;
    }
    acc *= 0.17677669529663687f;   // 1/sqrt(32)
    score[gid] = acc;
    atomicMax(&mU[(size_t)d * HEADS + h], orderF(acc));
}

// e = exp(score - max); segment sum via float atomics
__global__ void edge_expsum_kernel(const int* __restrict__ dst,
                                   float* __restrict__ score,
                                   const unsigned* __restrict__ mU,
                                   float* __restrict__ ssum, int EH) {
    int gid = blockIdx.x * blockDim.x + threadIdx.x;
    if (gid >= EH) return;
    int e = gid >> 3, h = gid & 7;
    int d = dst[e];
    float m = unorderF(mU[(size_t)d * HEADS + h]);
    if (!(m > -3.0e38f)) m = 0.0f;             // empty segment: -inf -> 0
    float ex = __expf(score[gid] - m);
    score[gid] = ex;
    atomicAdd(&ssum[(size_t)d * HEADS + h], ex);
}

// msg[dst] += V[src] * attn  (float atomics per head slice)
__global__ void edge_scatter_kernel(const float* __restrict__ V,
                                    const int* __restrict__ src, const int* __restrict__ dst,
                                    const float* __restrict__ score,
                                    const float* __restrict__ ssum,
                                    float* __restrict__ msg, int EH) {
    int gid = blockIdx.x * blockDim.x + threadIdx.x;
    if (gid >= EH) return;
    int e = gid >> 3, h = gid & 7;
    int s = src[e], d = dst[e];
    float w = score[gid] / (ssum[(size_t)d * HEADS + h] + 1e-16f);
    const float* vp = V   + (size_t)s * DIMC + h * HDIM;
    float*       op = msg + (size_t)d * DIMC + h * HDIM;
#pragma unroll
    for (int i = 0; i < HDIM; ++i)
        atomicAdd(op + i, vp[i] * w);
}

// ---------------------------------------------------------------------------
// Host-side branch driver
// ---------------------------------------------------------------------------
static inline int ceil_div(int a, int b) { return (a + b - 1) / b; }

static void run_branch(const float* x, const int* edges, int N, int E,
                       const float* wq, const float* bq, const float* wk, const float* bk,
                       const float* wv, const float* bv, const float* wo, const float* bo,
                       float* out, char* base, size_t& off, hipStream_t stream) {
    auto alloc = [&](size_t bytes) -> void* {
        void* p = base + off;
        off = (off + bytes + 255) & ~(size_t)255;
        return p;
    };

    int m_tiles = ceil_div(N, 16);
    size_t packA_halves = (size_t)m_tiles * 4096;

    _Float16* xA   = (_Float16*)alloc(packA_halves * 2);
    _Float16* wqB  = (_Float16*)alloc(65536 * 2);
    _Float16* wkB  = (_Float16*)alloc(65536 * 2);
    _Float16* wvB  = (_Float16*)alloc(65536 * 2);
    _Float16* woB  = (_Float16*)alloc(65536 * 2);
    float*    Q    = (float*)alloc((size_t)N * DIMC * 4);
    float*    K    = (float*)alloc((size_t)N * DIMC * 4);
    float*    V    = (float*)alloc((size_t)N * DIMC * 4);
    float*    scr  = (float*)alloc((size_t)E * HEADS * 4);
    unsigned* mU   = (unsigned*)alloc((size_t)N * HEADS * 4);
    float*    ssum = (float*)alloc((size_t)N * HEADS * 4);
    float*    msg  = (float*)alloc((size_t)N * DIMC * 4);
    _Float16* msgA = (_Float16*)alloc(packA_halves * 2);

    const int* src = edges;
    const int* dst = edges + E;

    // 1) pack weights and input
    pack_b_kernel<<<65536 / TB, TB, 0, stream>>>(wq, wqB);
    pack_b_kernel<<<65536 / TB, TB, 0, stream>>>(wk, wkB);
    pack_b_kernel<<<65536 / TB, TB, 0, stream>>>(wv, wvB);
    pack_b_kernel<<<65536 / TB, TB, 0, stream>>>(wo, woB);
    int totA = m_tiles << 12;
    pack_a_kernel<<<ceil_div(totA, TB), TB, 0, stream>>>(x, xA, N, m_tiles);

    // 2) Q, K, V projections (WMMA, B + bias staged in LDS)
    dim3 ggrid(ceil_div(m_tiles, 8));
    gemm_wmma_kernel<false><<<ggrid, TB, 0, stream>>>(xA, wqB, bq, nullptr, Q, N, m_tiles);
    gemm_wmma_kernel<false><<<ggrid, TB, 0, stream>>>(xA, wkB, bk, nullptr, K, N, m_tiles);
    gemm_wmma_kernel<false><<<ggrid, TB, 0, stream>>>(xA, wvB, bv, nullptr, V, N, m_tiles);

    // 3) segment softmax over edges + message aggregation
    int NH = N * HEADS, NC = N * DIMC, EH = E * HEADS;
    init_branch_kernel<<<ceil_div(NC, TB), TB, 0, stream>>>(mU, ssum, msg, NH, NC);
    edge_scores_kernel<<<ceil_div(EH, TB), TB, 0, stream>>>(Q, K, src, dst, scr, mU, EH);
    edge_expsum_kernel<<<ceil_div(EH, TB), TB, 0, stream>>>(dst, scr, mU, ssum, EH);
    edge_scatter_kernel<<<ceil_div(EH, TB), TB, 0, stream>>>(V, src, dst, scr, ssum, msg, EH);

    // 4) output projection + identity residual (WMMA)
    pack_a_kernel<<<ceil_div(totA, TB), TB, 0, stream>>>(msg, msgA, N, m_tiles);
    gemm_wmma_kernel<true><<<ggrid, TB, 0, stream>>>(msgA, woB, bo, x, out, N, m_tiles);
}

// ---------------------------------------------------------------------------
extern "C" void kernel_launch(void* const* d_in, const int* in_sizes, int n_in,
                              void* d_out, int out_size, void* d_ws, size_t ws_size,
                              hipStream_t stream) {
    const float* slot_x  = (const float*)d_in[0];
    const float* dom_x   = (const float*)d_in[1];
    const int*   slot_e  = (const int*)d_in[2];
    const int*   dom_e   = (const int*)d_in[3];

    int N_s = in_sizes[0] / DIMC;
    int N_d = in_sizes[1] / DIMC;
    int E_s = in_sizes[2] / 2;
    int E_d = in_sizes[3] / 2;

    float* out_slot = (float*)d_out;
    float* out_dom  = out_slot + (size_t)N_s * DIMC;

    char*  base = (char*)d_ws;
    size_t off  = 0;

    run_branch(slot_x, slot_e, N_s, E_s,
               (const float*)d_in[4],  (const float*)d_in[5],
               (const float*)d_in[6],  (const float*)d_in[7],
               (const float*)d_in[8],  (const float*)d_in[9],
               (const float*)d_in[10], (const float*)d_in[11],
               out_slot, base, off, stream);

    run_branch(dom_x, dom_e, N_d, E_d,
               (const float*)d_in[12], (const float*)d_in[13],
               (const float*)d_in[14], (const float*)d_in[15],
               (const float*)d_in[16], (const float*)d_in[17],
               (const float*)d_in[18], (const float*)d_in[19],
               out_dom, base, off, stream);
}
